// AttentionLayer_81630148428339
// MI455X (gfx1250) — matrix operand order repaired and checked by reading, not compiled, not run on hardware
//
#include <hip/hip_runtime.h>

// ---------------------------------------------------------------------------
// Attention layer for MI455X (gfx1250, wave32, WMMA).
// All matmuls via v_wmma_f32_16x16x32_bf16 (fp32 accumulate).
// ~137 GFLOPs vs ~100MB HBM traffic -> compute bound; bf16 operands, K/V are
// L2-resident (32MB << 192MB).  Attention computes S^T = K x Q^T with a
// permuted key-row order so exp(S) packs in-register directly into the
// B-operand of ctx^T = V^T x P^T: zero LDS traffic, 2 shuffles per subtile.
// Each wave processes 32 queries (2 subtiles) so every K/V fragment load
// feeds two WMMAs: 32 FLOPs per L2 byte in the attention core.
// ---------------------------------------------------------------------------

typedef __attribute__((ext_vector_type(16))) __bf16 v16bf;
typedef __attribute__((ext_vector_type(8)))  float  v8f;

#define Bb  4
#define Ss  2048
#define Dd  1024
#define Hh  16
#define HDd 64

// fp32 -> bf16 round-to-nearest-even (integer ALU only)
static __device__ __forceinline__ unsigned short f2bf(float f) {
    unsigned u = __float_as_uint(f);
    u += 0x7FFFu + ((u >> 16) & 1u);
    return (unsigned short)(u >> 16);
}

union FragBF  { v16bf v; uint4 u[2]; unsigned short s[16]; };
union PackB64 { uint4 u; unsigned short s[8]; };

// A-operand fragment: 16(M) x 32(K) bf16 from row-major src (K contiguous).
// Layout: lanes 0-15 row M=lane, K=0..15; lanes 16-31 K=16..31.
static __device__ __forceinline__ v16bf loadAfrag(const unsigned short* src, int ld) {
    int lane = threadIdx.x & 31;
    const unsigned short* p = src + (lane & 15) * ld + (lane >> 4) * 16;
    FragBF f;
    f.u[0] = *(const uint4*)(p);
    f.u[1] = *(const uint4*)(p + 8);
    return f.v;
}

// Same A-frag but with permuted row mapping (row -> row + (row&8)), used for
// the K-tiles so score rows land at keys {0..7,16..23 | 8..15,24..31}: this
// makes the score C-frag convert directly into a B-operand fragment.
static __device__ __forceinline__ v16bf loadAfragPerm(const unsigned short* src, int k0) {
    int lane = threadIdx.x & 31;
    int lrow = lane & 15;
    int prow = lrow + (lrow & 8);
    const unsigned short* p = src + prow * HDd + k0 + (lane >> 4) * 16;
    FragBF f;
    f.u[0] = *(const uint4*)(p);
    f.u[1] = *(const uint4*)(p + 8);
    return f.v;
}

// B-operand via Y^T trick: Y is 16(N) x 32(K) row-major -> B = Y^T.
// Layout: lane n holds col n; lanes 0-15 K=0..15, lanes 16-31 K=16..31.
static __device__ __forceinline__ v16bf loadBTfrag(const unsigned short* src, int ld) {
    int lane = threadIdx.x & 31;
    const unsigned short* p = src + (lane & 15) * ld + (lane >> 4) * 16;
    FragBF f;
    f.u[0] = *(const uint4*)(p);
    f.u[1] = *(const uint4*)(p + 8);
    return f.v;
}

static __device__ __forceinline__ v8f wmma_bf16(v16bf a, v16bf b, v8f c) {
    return __builtin_amdgcn_wmma_f32_16x16x32_bf16(false, a, false, b,
                                                   (short)0, c, false, false);
}

// ---------------------------------------------------------------------------
// Prep kernels (memory-trivial)
// ---------------------------------------------------------------------------
__global__ void __launch_bounds__(256)
conv_bf(const float* __restrict__ X, unsigned short* __restrict__ Y, int n) {
    int i = blockIdx.x * blockDim.x + threadIdx.x;
    if (i < n) Y[i] = f2bf(X[i]);
}

// Wt[n][k] = bf16(W[k][n])
__global__ void __launch_bounds__(256)
transpose_w(const float* __restrict__ W, unsigned short* __restrict__ Wt) {
    int idx = blockIdx.x * blockDim.x + threadIdx.x;   // idx = n*D + k
    int n = idx >> 10, k = idx & (Dd - 1);
    Wt[idx] = f2bf(W[k * Dd + n]);
}

// ---------------------------------------------------------------------------
// WMMA GEMM: C[8192 x 1024] = A(bf16 row-major) x Wt^T.  One wave = 32x64 tile,
// 8 WMMAs per K-step, 256 WMMAs/block (21 FLOPs per L2 byte).
// MODE 0: bf16 -> [B,H,S,HD]   (Q, K)
// MODE 1: bf16 -> [B,H,HD,S]   (V transposed; packed b128 stores)
// MODE 2: fp32 -> [M, D]       (final output)
// ---------------------------------------------------------------------------
template <int MODE>
__global__ void __launch_bounds__(32)
gemm_ws(const unsigned short* __restrict__ A,
        const unsigned short* __restrict__ Wt,
        unsigned short* __restrict__ outb,
        float* __restrict__ outf) {
    const int M = Bb * Ss;
    int mt = blockIdx.x % (M / 32);
    int nt = blockIdx.x / (M / 32);
    int row0 = mt * 32, n0 = nt * 64;

    v8f acc[2][4];
    #pragma unroll
    for (int r = 0; r < 2; ++r)
        #pragma unroll
        for (int i = 0; i < 4; ++i)
            acc[r][i] = (v8f){0.f,0.f,0.f,0.f,0.f,0.f,0.f,0.f};

    for (int k0 = 0; k0 < Dd; k0 += 32) {
        v16bf a0 = loadAfrag(A + (size_t)row0        * Dd + k0, Dd);
        v16bf a1 = loadAfrag(A + (size_t)(row0 + 16) * Dd + k0, Dd);
        #pragma unroll
        for (int i = 0; i < 4; ++i) {
            v16bf y = loadBTfrag(Wt + (size_t)(n0 + 16 * i) * Dd + k0, Dd);
            acc[0][i] = wmma_bf16(a0, y, acc[0][i]);
            acc[1][i] = wmma_bf16(a1, y, acc[1][i]);
        }
    }

    int lane = threadIdx.x & 31;
    int nl = lane & 15, half = lane >> 4;
    #pragma unroll
    for (int rt = 0; rt < 2; ++rt) {
        #pragma unroll
        for (int i = 0; i < 4; ++i) {
            int n = n0 + i * 16 + nl;
            if (MODE == 1) {
                // s varies with r (consecutive) -> pack 8 bf16 into one b128
                int m0 = row0 + rt * 16 + 8 * half;
                int b = m0 >> 11, sb = m0 & (Ss - 1);
                int h = n >> 6, hd = n & (HDd - 1);
                PackB64 pk;
                #pragma unroll
                for (int r = 0; r < 8; ++r) pk.s[r] = f2bf(acc[rt][i][r]);
                *(uint4*)(outb + ((size_t)(b * Hh + h) * HDd + hd) * Ss + sb) = pk.u;
            } else {
                #pragma unroll
                for (int r = 0; r < 8; ++r) {
                    int m = row0 + rt * 16 + r + 8 * half;   // C-layout M
                    float v = acc[rt][i][r];
                    if (MODE == 2) {
                        outf[(size_t)m * Dd + n] = v;
                    } else {
                        int b = m >> 11, s = m & (Ss - 1);
                        int h = n >> 6,  hd = n & (HDd - 1);
                        outb[(((size_t)(b * Hh + h)) * Ss + s) * HDd + hd] = f2bf(v);
                    }
                }
            }
        }
    }
}

// ---------------------------------------------------------------------------
// Flash attention, one wave per (b,h,32-query tile = 2 subtiles).
//   S^T = Kperm (A) x Q^T (B)   -> query = lane column -> in-lane softmax
//   P packs in-register into B-operand of  ctx^T = V^T (A) x P^T (B)
// Per 32-key chunk: 8 score WMMAs + 8 ctx WMMAs from one 8KB K/V tile load.
// ---------------------------------------------------------------------------
__global__ void __launch_bounds__(32)
attn_fa(const unsigned short* __restrict__ Qb,
        const unsigned short* __restrict__ Kb,
        const unsigned short* __restrict__ Vt,
        unsigned short* __restrict__ Ctx) {
    const int TQ = Ss / 32;
    int t  = blockIdx.x % TQ;
    int bh = blockIdx.x / TQ;
    int b = bh >> 4, h = bh & (Hh - 1);

    const unsigned short* Qh = Qb + (size_t)bh * Ss * HDd;
    const unsigned short* Kh = Kb + (size_t)bh * Ss * HDd;
    const unsigned short* Vh = Vt + (size_t)bh * HDd * Ss;

    int lane = threadIdx.x & 31;
    int nl = lane & 15, half = lane >> 4;
    int qbase = t * 32;

    // Q^T B-fragments: 2 query subtiles x 2 hd chunks, loaded once
    v16bf qb00 = loadBTfrag(Qh + (size_t)qbase * HDd,             HDd);
    v16bf qb01 = loadBTfrag(Qh + (size_t)qbase * HDd + 32,        HDd);
    v16bf qb10 = loadBTfrag(Qh + (size_t)(qbase + 16) * HDd,      HDd);
    v16bf qb11 = loadBTfrag(Qh + (size_t)(qbase + 16) * HDd + 32, HDd);

    v8f acc[2][4];                             // ctx^T per subtile
    #pragma unroll
    for (int s = 0; s < 2; ++s)
        #pragma unroll
        for (int i = 0; i < 4; ++i)
            acc[s][i] = (v8f){0.f,0.f,0.f,0.f,0.f,0.f,0.f,0.f};
    float mr[2] = {-1e30f, -1e30f}, lr[2] = {0.f, 0.f};
    int qi[2] = {qbase + nl, qbase + 16 + nl}; // this lane's query per subtile

    const float sc = 0.125f * 1.44269504088896340736f;   // rsqrt(64)*log2(e)

    for (int kc = 0; kc <= qbase + 31; kc += 32) {
        // K fragments loaded once, reused by both query subtiles
        v16bf ka00 = loadAfragPerm(Kh + (size_t)kc * HDd,       0);
        v16bf ka01 = loadAfragPerm(Kh + (size_t)kc * HDd,       32);
        v16bf ka10 = loadAfragPerm(Kh + (size_t)(kc + 8) * HDd, 0);
        v16bf ka11 = loadAfragPerm(Kh + (size_t)(kc + 8) * HDd, 32);

        bool needMask = (kc + 31 > qbase);     // only the diagonal chunk
        int kb = kc + 16 * half;               // this lane's first key
        FragBF pb[2];
        float alpha[2];

        #pragma unroll
        for (int s = 0; s < 2; ++s) {
            v16bf q0 = s ? qb10 : qb00;
            v16bf q1 = s ? qb11 : qb01;
            v8f st0 = (v8f){0.f,0.f,0.f,0.f,0.f,0.f,0.f,0.f};
            v8f st1 = st0;
            st0 = wmma_bf16(ka00, q0, st0);
            st0 = wmma_bf16(ka01, q1, st0);
            st1 = wmma_bf16(ka10, q0, st1);
            st1 = wmma_bf16(ka11, q1, st1);

            // lane holds keys kb+{0..7} (st0), kb+{8..15} (st1) for query qi[s]
            float z0[8], z1[8];
            #pragma unroll
            for (int r = 0; r < 8; ++r) {
                z0[r] = st0[r] * sc;
                z1[r] = st1[r] * sc;
                if (needMask) {
                    if (kb + r     > qi[s]) z0[r] = -1e30f;
                    if (kb + 8 + r > qi[s]) z1[r] = -1e30f;
                }
            }
            // query max: in-lane over 16 values + one cross-half shuffle
            float rm = fmaxf(z0[0], z1[0]);
            #pragma unroll
            for (int r = 1; r < 8; ++r) rm = fmaxf(rm, fmaxf(z0[r], z1[r]));
            rm = fmaxf(rm, __shfl_xor(rm, 16, 32));
            float mnew = fmaxf(mr[s], rm);
            alpha[s] = exp2f(mr[s] - mnew);

            // P = exp2(z - mnew), packed straight into the B-operand frag:
            // element j <-> chunk-local key 16*half + j (key permutation)
            float ps = 0.f;
            #pragma unroll
            for (int r = 0; r < 8; ++r) {
                float p0 = exp2f(z0[r] - mnew);
                float p1 = exp2f(z1[r] - mnew);
                ps += p0 + p1;
                pb[s].s[r]     = f2bf(p0);
                pb[s].s[8 + r] = f2bf(p1);
            }
            ps += __shfl_xor(ps, 16, 32);
            lr[s] = lr[s] * alpha[s] + ps;
            mr[s] = mnew;
        }

        // ctx^T += V^T x P^T : V fragments loaded once, feed both subtiles
        #pragma unroll
        for (int i = 0; i < 4; ++i) {
            v16bf va = loadAfrag(Vh + (size_t)(i * 16) * Ss + kc, Ss);
            acc[0][i] = wmma_bf16(va, pb[0].v, acc[0][i] * alpha[0]);
            acc[1][i] = wmma_bf16(va, pb[1].v, acc[1][i] * alpha[1]);
        }
    }

    // normalize, write ctx rows: 8 consecutive hd per frag -> b128 stores
    #pragma unroll
    for (int s = 0; s < 2; ++s) {
        float inv = 1.0f / lr[s];
        size_t rowoff = ((size_t)(b * Ss + qi[s])) * Dd + h * HDd + 8 * half;
        #pragma unroll
        for (int i = 0; i < 4; ++i) {
            PackB64 pk;
            #pragma unroll
            for (int r = 0; r < 8; ++r) pk.s[r] = f2bf(acc[s][i][r] * inv);
            *(uint4*)(Ctx + rowoff + i * 16) = pk.u;
        }
    }
}

// ---------------------------------------------------------------------------
extern "C" void kernel_launch(void* const* d_in, const int* in_sizes, int n_in,
                              void* d_out, int out_size, void* d_ws, size_t ws_size,
                              hipStream_t stream) {
    const float* states = (const float*)d_in[0];
    // d_in[1] = causal mask (implied by kernel, unused)
    const float* Wq = (const float*)d_in[2];
    const float* Wk = (const float*)d_in[3];
    const float* Wv = (const float*)d_in[4];
    const float* Wo = (const float*)d_in[5];
    float* out = (float*)d_out;

    // Workspace layout (88 MB, every region fully rewritten each call)
    const size_t MB = (size_t)1 << 20;
    char* w = (char*)d_ws;
    unsigned short* Xbf = (unsigned short*)(w);             // 16 MB states bf16
    unsigned short* Wtq = (unsigned short*)(w + 16 * MB);   //  2 MB
    unsigned short* Wtk = (unsigned short*)(w + 18 * MB);   //  2 MB
    unsigned short* Wtv = (unsigned short*)(w + 20 * MB);   //  2 MB
    unsigned short* Wto = (unsigned short*)(w + 22 * MB);   //  2 MB
    unsigned short* Qb  = (unsigned short*)(w + 24 * MB);   // 16 MB [B,H,S,HD]
    unsigned short* Kb  = (unsigned short*)(w + 40 * MB);   // 16 MB [B,H,S,HD]
    unsigned short* Vtb = (unsigned short*)(w + 56 * MB);   // 16 MB [B,H,HD,S]
    unsigned short* Ctx = (unsigned short*)(w + 72 * MB);   // 16 MB [B*S,D]

    const int nX = Bb * Ss * Dd;
    conv_bf<<<nX / 256, 256, 0, stream>>>(states, Xbf, nX);
    transpose_w<<<(Dd * Dd) / 256, 256, 0, stream>>>(Wq, Wtq);
    transpose_w<<<(Dd * Dd) / 256, 256, 0, stream>>>(Wk, Wtk);
    transpose_w<<<(Dd * Dd) / 256, 256, 0, stream>>>(Wv, Wtv);
    transpose_w<<<(Dd * Dd) / 256, 256, 0, stream>>>(Wo, Wto);

    dim3 ggrid((Bb * Ss / 32) * (Dd / 64));                 // 4096 waves
    gemm_ws<0><<<ggrid, 32, 0, stream>>>(Xbf, Wtq, Qb,  nullptr);
    gemm_ws<0><<<ggrid, 32, 0, stream>>>(Xbf, Wtk, Kb,  nullptr);
    gemm_ws<1><<<ggrid, 32, 0, stream>>>(Xbf, Wtv, Vtb, nullptr);

    attn_fa<<<Bb * Hh * (Ss / 32), 32, 0, stream>>>(Qb, Kb, Vtb, Ctx);

    gemm_ws<2><<<ggrid, 32, 0, stream>>>(Ctx, Wto, nullptr, out);
}
